// SparseSAE_12429635355389
// MI455X (gfx1250) — matrix operand order repaired and checked by reading, not compiled, not run on hardware
//
#include <hip/hip_runtime.h>

// ---------------------------------------------------------------------------
// Sparse autoencoder forward for MI455X (gfx1250, wave32, WMMA).
//   z_pre = x @ enc_w^T + enc_b            (bf16 WMMA, f32 accum, pipelined)
//   z     = scatter(relu(top32(z_pre)))    (exact radix select, in place)
//   x_hat = z @ dec_w^T + dec_b            (sparse gather, L2-resident dec_w)
// ---------------------------------------------------------------------------

typedef __bf16 bf16_t;
typedef __attribute__((ext_vector_type(16))) __bf16   v16bf;
typedef __attribute__((ext_vector_type(8)))  float    v8f;
typedef __attribute__((ext_vector_type(4)))  unsigned u32x4;
typedef __attribute__((ext_vector_type(2)))  unsigned u32x2;

#define NROWS 8192
#define CDIM  1024
#define KDIM  16384
#define TOPK  32

#define TM 128
#define TN 128
#define TK 32
#define KITERS (CDIM / TK)   // 32
#define LDP 40               // LDS row pitch in bf16 (80B: bank-spread, 16B aligned)
#define LDSZ (TM * LDP)      // one buffer, elements

// ---------------------------------------------------------------------------
// Kernel 1: z_pre = x @ enc_w^T + enc_b   via v_wmma_f32_16x16x32_bf16
// 256 threads (8 waves, 2x4 wave grid), tile 128x128, K-step 32,
// double-buffered LDS: global loads for slice k+1 overlap the 8 WMMAs of k.
// ---------------------------------------------------------------------------
__global__ __launch_bounds__(256) void sae_enc_gemm(
    const float* __restrict__ x,      // [NROWS][CDIM]
    const float* __restrict__ w,      // [KDIM][CDIM]
    const float* __restrict__ bias,   // [KDIM]
    float* __restrict__ zpre)         // [NROWS][KDIM]
{
    __shared__ bf16_t As[2 * LDSZ];
    __shared__ bf16_t Bs[2 * LDSZ];

    const int tid  = threadIdx.x;
    const int lane = tid & 31;
    const int wave = tid >> 5;       // 0..7
    const int wm   = wave & 1;       // 2 waves along M
    const int wn   = wave >> 1;      // 4 waves along N
    const int bm   = blockIdx.x * TM;
    const int bn   = blockIdx.y * TN;

    v8f acc[4][2];
#pragma unroll
    for (int mi = 0; mi < 4; ++mi)
#pragma unroll
        for (int ni = 0; ni < 2; ++ni)
            acc[mi][ni] = (v8f){0.f, 0.f, 0.f, 0.f, 0.f, 0.f, 0.f, 0.f};

    // staging map: 256 threads cover a 128x32 f32 tile; thread -> (row, 4 floats)
    const int srow = tid >> 3;           // 0..31 (4 sweeps of 32 rows)
    const int skq  = (tid & 7) * 4;      // k offset in floats

    const int rlow = lane & 15;
    const int rsel = lane >> 4;          // lane half selects K sub-range

    const float* xg = x + (size_t)(bm + srow) * CDIM + skq;
    const float* wg = w + (size_t)(bn + srow) * CDIM + skq;

    // prologue: global loads for slice 0
    float4 ar[4], br[4];
#pragma unroll
    for (int s = 0; s < 4; ++s) {
        ar[s] = *(const float4*)(xg + (size_t)(s * 32) * CDIM);
        br[s] = *(const float4*)(wg + (size_t)(s * 32) * CDIM);
    }

    for (int it = 0; it < KITERS; ++it) {
        const int buf = it & 1;
        bf16_t* Asb = As + buf * LDSZ;
        bf16_t* Bsb = Bs + buf * LDSZ;

        // convert staged registers -> LDS (buf)
#pragma unroll
        for (int s = 0; s < 4; ++s) {
            const int r = srow + s * 32;
            union { bf16_t h[4]; u32x2 u; } pa, pb;
            pa.h[0] = (bf16_t)ar[s].x; pa.h[1] = (bf16_t)ar[s].y;
            pa.h[2] = (bf16_t)ar[s].z; pa.h[3] = (bf16_t)ar[s].w;
            pb.h[0] = (bf16_t)br[s].x; pb.h[1] = (bf16_t)br[s].y;
            pb.h[2] = (bf16_t)br[s].z; pb.h[3] = (bf16_t)br[s].w;
            *(u32x2*)&Asb[r * LDP + skq] = pa.u;
            *(u32x2*)&Bsb[r * LDP + skq] = pb.u;
        }
        __syncthreads();   // buf ready; buf^1 free (its readers finished last epoch)

        // issue global loads for next K-slice: latency hides under the WMMAs
        if (it + 1 < KITERS) {
            const int kt = (it + 1) * TK;
#pragma unroll
            for (int s = 0; s < 4; ++s) {
                ar[s] = *(const float4*)(xg + (size_t)(s * 32) * CDIM + kt);
                br[s] = *(const float4*)(wg + (size_t)(s * 32) * CDIM + kt);
            }
            if (it + 2 < KITERS) {   // global_prefetch hint two slices ahead
                __builtin_prefetch(xg + kt + TK, 0, 1);
                __builtin_prefetch(wg + kt + TK, 0, 1);
            }
        }

        // A fragment (16x32 bf16): lanes 0-15 -> K{0..7,16..23}, 16-31 -> K{8..15,24..31}
        v16bf afr[4];
#pragma unroll
        for (int mi = 0; mi < 4; ++mi) {
            const int row = wm * 64 + mi * 16 + rlow;
            const char* p = (const char*)Asb + row * (LDP * 2) + rsel * 16;
            union { u32x4 q[2]; v16bf v; } t;
            t.q[0] = *(const u32x4*)(p);
            t.q[1] = *(const u32x4*)(p + 32);
            afr[mi] = t.v;
        }
        // B fragment (32x16 bf16): lanes 0-15 -> K{0..7,16..23}, 16-31 -> K{8..15,24..31}
        v16bf bfr[2];
#pragma unroll
        for (int ni = 0; ni < 2; ++ni) {
            const int row = wn * 32 + ni * 16 + rlow;
            const char* p = (const char*)Bsb + row * (LDP * 2) + rsel * 16;
            union { u32x4 q[2]; v16bf v; } t;
            t.q[0] = *(const u32x4*)(p);
            t.q[1] = *(const u32x4*)(p + 32);
            bfr[ni] = t.v;
        }
#pragma unroll
        for (int mi = 0; mi < 4; ++mi)
#pragma unroll
            for (int ni = 0; ni < 2; ++ni)
                acc[mi][ni] = __builtin_amdgcn_wmma_f32_16x16x32_bf16(
                    false, afr[mi], false, bfr[ni],
                    (short)0, acc[mi][ni], false, false);
        // no trailing barrier: next iteration stores to buf^1 (disjoint),
        // and buf is only overwritten after one more barrier.
    }

    // epilogue: C/D layout — lane%16 = N col; vgpr r = row r (lanes 0-15) / r+8 (16-31)
    const int mhalf = rsel * 8;
#pragma unroll
    for (int mi = 0; mi < 4; ++mi)
#pragma unroll
        for (int ni = 0; ni < 2; ++ni) {
            const int gn = bn + wn * 32 + ni * 16 + rlow;
            const float b = bias[gn];
#pragma unroll
            for (int r = 0; r < 8; ++r) {
                const int gm = bm + wm * 64 + mi * 16 + mhalf + r;
                zpre[(size_t)gm * KDIM + gn] = acc[mi][ni][r] + b;
            }
        }
}

// ---------------------------------------------------------------------------
// Kernel 2: exact top-32 per row, relu, in-place sparsify.
// 4-round radix select on monotone sortable keys; row stays L2-hot.
// ---------------------------------------------------------------------------
__global__ __launch_bounds__(256) void sae_topk(float* __restrict__ z)
{
    const int row = blockIdx.x;
    float* zr = z + (size_t)row * KDIM;
    const int tid = threadIdx.x;

    __shared__ unsigned hist[256];
    __shared__ unsigned s_pf;
    __shared__ int s_r;
    __shared__ int eq_idx[256];
    __shared__ unsigned eq_n;

    if (tid == 0) { s_pf = 0u; s_r = TOPK; eq_n = 0u; }

    for (int round = 0; round < 4; ++round) {
        hist[tid] = 0u;
        __syncthreads();
        const unsigned pf = s_pf;
        const int nb = round * 8;
        for (int i = tid; i < KDIM; i += 256) {
            const unsigned u = __float_as_uint(zr[i]);
            const unsigned k = u ^ (((unsigned)((int)u >> 31)) | 0x80000000u);
            if (nb == 0 || (k >> (32 - nb)) == pf)
                atomicAdd(&hist[(k >> (24 - nb)) & 0xFFu], 1u);
        }
        __syncthreads();
        if (tid == 0) {
            const int r = s_r;
            unsigned cum = 0; int bin = 0;
            for (int b = 255; b >= 0; --b) {
                const unsigned c = hist[b];
                if (cum + c >= (unsigned)r) { bin = b; s_r = r - (int)cum; break; }
                cum += c;
            }
            s_pf = (s_pf << 8) | (unsigned)bin;
        }
        __syncthreads();
    }

    const unsigned t = s_pf;   // exact key of the 32nd-largest value
    const int need = s_r;      // how many key==t elements to keep (lowest index first)

    for (int i = tid; i < KDIM; i += 256) {
        const float v = zr[i];
        const unsigned u = __float_as_uint(v);
        const unsigned k = u ^ (((unsigned)((int)u >> 31)) | 0x80000000u);
        float out = 0.f;
        if (k > t) {
            out = v > 0.f ? v : 0.f;
        } else if (k == t) {
            const unsigned p = atomicAdd(&eq_n, 1u);
            if (p < 256u) eq_idx[p] = i;
        }
        zr[i] = out;
    }
    __syncthreads();
    if (tid == 0) {
        int n = (eq_n < 256u) ? (int)eq_n : 256;
        for (int a = 1; a < n; ++a) {            // tiny insertion sort by index
            const int v = eq_idx[a]; int b = a - 1;
            while (b >= 0 && eq_idx[b] > v) { eq_idx[b + 1] = eq_idx[b]; --b; }
            eq_idx[b + 1] = v;
        }
        int kc = need < n ? need : n;
        if (kc < 0) kc = 0;
        // recover the tied value from its key (involution inverse), then relu
        const unsigned ut = (t & 0x80000000u) ? (t ^ 0x80000000u) : ~t;
        float vt = __uint_as_float(ut);
        if (!(vt > 0.f)) vt = 0.f;
        for (int j = 0; j < kc; ++j) zr[eq_idx[j]] = vt;
    }
}

// ---------------------------------------------------------------------------
// Kernel 3: x_hat[row] = sum_j z_j * dec_w[:, idx_j] + dec_b  (sparse decode)
// dec_w (64MB) is L2-resident; strided column gathers hit L2, not HBM.
// ---------------------------------------------------------------------------
__global__ __launch_bounds__(256) void sae_decode(
    const float* __restrict__ z,      // [NROWS][KDIM], sparse (>=0)
    const float* __restrict__ dec_w,  // [CDIM][KDIM]
    const float* __restrict__ dec_b,  // [CDIM]
    float* __restrict__ xhat)         // [NROWS][CDIM]
{
    const int row = blockIdx.x;
    const int tid = threadIdx.x;
    __shared__ float sv[64];
    __shared__ int   si[64];
    __shared__ unsigned sn;
    if (tid == 0) sn = 0u;
    __syncthreads();

    const float* zr = z + (size_t)row * KDIM;
    for (int i = tid; i < KDIM; i += 256) {
        const float v = zr[i];
        if (v > 0.f) {
            const unsigned p = atomicAdd(&sn, 1u);
            if (p < 64u) { sv[p] = v; si[p] = i; }
        }
    }
    __syncthreads();
    const int nn = (sn < 64u) ? (int)sn : 64;

    for (int c = tid; c < CDIM; c += 256) {
        float acc = dec_b[c];
        const float* wr = dec_w + (size_t)c * KDIM;
        for (int j = 0; j < nn; ++j) acc += sv[j] * wr[si[j]];
        xhat[(size_t)row * CDIM + c] = acc;
    }
}

// ---------------------------------------------------------------------------
extern "C" void kernel_launch(void* const* d_in, const int* in_sizes, int n_in,
                              void* d_out, int out_size, void* d_ws, size_t ws_size,
                              hipStream_t stream) {
    (void)in_sizes; (void)n_in; (void)out_size; (void)d_ws; (void)ws_size;
    const float* x     = (const float*)d_in[0];
    const float* enc_w = (const float*)d_in[1];
    const float* enc_b = (const float*)d_in[2];
    const float* dec_w = (const float*)d_in[3];
    const float* dec_b = (const float*)d_in[4];

    float* z    = (float*)d_out;                    // [NROWS][KDIM]
    float* xhat = z + (size_t)NROWS * KDIM;         // [NROWS][CDIM]

    dim3 grid(NROWS / TM, KDIM / TN);
    sae_enc_gemm<<<grid, 256, 0, stream>>>(x, enc_w, enc_b, z);
    sae_topk<<<NROWS, 256, 0, stream>>>(z);
    sae_decode<<<NROWS, 256, 0, stream>>>(z, dec_w, dec_b, xhat);
}
